// MultiHeadSelfAttention_37211596653207
// MI455X (gfx1250) — matrix operand order repaired and checked by reading, not compiled, not run on hardware
//
#include <hip/hip_runtime.h>
#include <hip/hip_bf16.h>
#include <math.h>

typedef __attribute__((ext_vector_type(16))) _Float16 v16h;
typedef __attribute__((ext_vector_type(8)))  _Float16 v8h;
typedef __attribute__((ext_vector_type(8)))  float    v8f;

union V16U { v16h v; v8h h[2]; };

#define WMMA_F32_F16(A, B, C) \
  __builtin_amdgcn_wmma_f32_16x16x32_f16(false, (A), false, (B), (short)0, (C), false, false)

static constexpr int Bb = 4, Ll = 2048, Dd = 1024, Hh = 16, DH = 64;
static constexpr int Mm = Bb * Ll;   // 8192

// ---------------------------------------------------------------------------
// Async global->LDS copy (CDNA5 GLOBAL_LOAD_ASYNC_TO_LDS_B128, ASYNCcnt).
// ---------------------------------------------------------------------------
#if __has_builtin(__builtin_amdgcn_global_load_async_to_lds_b128)
#define HAS_ASYNC_LDS 1
typedef int vec4i __attribute__((vector_size(16)));
typedef __attribute__((address_space(1))) vec4i gvec4i;
typedef __attribute__((address_space(3))) vec4i lvec4i;
__device__ __forceinline__ void cp16_async(void* lds, const void* g) {
  __builtin_amdgcn_global_load_async_to_lds_b128((gvec4i*)g, (lvec4i*)lds, 0, 0);
}
#else
#define HAS_ASYNC_LDS 0
__device__ __forceinline__ void cp16_async(void* lds, const void* g) {
  *(int4*)lds = *(const int4*)g;   // synchronous fallback
}
#endif

__device__ __forceinline__ void wait_async_le4() {
#if __has_builtin(__builtin_amdgcn_s_wait_asynccnt)
  __builtin_amdgcn_s_wait_asynccnt(4);
#else
  asm volatile("s_wait_asynccnt 4" ::: "memory");
#endif
}

// ---------------------------------------------------------------------------
// f32 -> f16 conversion (grid-stride)
// ---------------------------------------------------------------------------
__global__ __launch_bounds__(256) void cvt_f32_to_f16_kernel(
    const float* __restrict__ in, _Float16* __restrict__ out, int n) {
  for (int i = blockIdx.x * blockDim.x + threadIdx.x; i < n;
       i += gridDim.x * blockDim.x)
    out[i] = (_Float16)in[i];
}

// ---------------------------------------------------------------------------
// GEMM: out[m,n] = sum_k A[m,k] * W[n,k] + bias[n]   (A: MxK f16, W: NxK f16)
// Wave tile 32x64 (2x4 WMMA accs, 8 WMMAs / k-step), block = 4 waves (64x128).
// 2-stage register pipeline: fragments for k+32 load while k is computed.
// MODE 0: f16 row-major  MODE 1: f16 per-head-transposed (Vt)  MODE 2: f32
// ---------------------------------------------------------------------------
template <int MODE>
__global__ __launch_bounds__(128) void gemm_wmma_kernel(
    const _Float16* __restrict__ A, const _Float16* __restrict__ W,
    const float* __restrict__ bias, void* __restrict__ outp,
    int M, int N, int K) {
  const int lane = threadIdx.x & 31;
  const int wid  = threadIdx.x >> 5;
  const int l15  = lane & 15;
  const int hi   = lane >> 4;
  const int m0   = blockIdx.y * 64  + (wid >> 1) * 32;
  const int n0   = blockIdx.x * 128 + (wid & 1) * 64;

  v8f acc[2][4] = {};

  const _Float16* arow[2];
  arow[0] = A + (size_t)(m0 + l15) * K;
  arow[1] = arow[0] + (size_t)16 * K;
  const _Float16* brow[4];
  brow[0] = W + (size_t)(n0 + l15) * K;
  brow[1] = brow[0] + (size_t)16 * K;
  brow[2] = brow[0] + (size_t)32 * K;
  brow[3] = brow[0] + (size_t)48 * K;
  const int aoff = hi * 16;
  const int boff = hi * 16;

  V16U a[2];
  v16h b[4];
#pragma unroll
  for (int i = 0; i < 2; ++i) {
    a[i].h[0] = *(const v8h*)(arow[i] + aoff);
    a[i].h[1] = *(const v8h*)(arow[i] + 8 + aoff);
  }
#pragma unroll
  for (int j = 0; j < 4; ++j) b[j] = *(const v16h*)(brow[j] + boff);

  for (int k0 = 0; k0 < K; k0 += 32) {
    const int kn = (k0 + 32 < K) ? k0 + 32 : k0;  // clamp: last reload is dead
    V16U na[2];
    v16h nb[4];
#pragma unroll
    for (int i = 0; i < 2; ++i) {
      na[i].h[0] = *(const v8h*)(arow[i] + kn + aoff);
      na[i].h[1] = *(const v8h*)(arow[i] + kn + 8 + aoff);
    }
#pragma unroll
    for (int j = 0; j < 4; ++j) nb[j] = *(const v16h*)(brow[j] + kn + boff);

#pragma unroll
    for (int i = 0; i < 2; ++i)
#pragma unroll
      for (int j = 0; j < 4; ++j)
        acc[i][j] = WMMA_F32_F16(a[i].v, b[j], acc[i][j]);

    a[0] = na[0]; a[1] = na[1];
    b[0] = nb[0]; b[1] = nb[1]; b[2] = nb[2]; b[3] = nb[3];
  }

  const int roff = hi * 8;  // C layout: vgpr r -> row roff+r, col l15
#pragma unroll
  for (int i = 0; i < 2; ++i) {
#pragma unroll
    for (int j = 0; j < 4; ++j) {
      const int n  = n0 + j * 16 + l15;
      const float bv = bias[n];
      if (MODE == 2) {
        float* O = (float*)outp;
#pragma unroll
        for (int r = 0; r < 8; ++r)
          O[(size_t)(m0 + i * 16 + roff + r) * N + n] = acc[i][j][r] + bv;
      } else if (MODE == 0) {
        _Float16* O = (_Float16*)outp;
#pragma unroll
        for (int r = 0; r < 8; ++r)
          O[(size_t)(m0 + i * 16 + roff + r) * N + n] =
              (_Float16)(acc[i][j][r] + bv);
      } else {  // MODE 1: Vt[((b*H + h)*DH + d) * L + l],  n == h*DH + d
        const int mb    = m0 + i * 16;
        const int batch = mb >> 11;
        const int lbase = (mb + roff) & (Ll - 1);
        _Float16* O = (_Float16*)outp + (size_t)batch * (Hh * DH * Ll) +
                      (size_t)n * Ll + lbase;
        v8h pk;
#pragma unroll
        for (int r = 0; r < 8; ++r) pk[r] = (_Float16)(acc[i][j][r] + bv);
        *(v8h*)O = pk;
      }
    }
  }
}

// ---------------------------------------------------------------------------
// Flash attention. Block = 4 waves, one (b,h), 64 queries (16/wave).
// K/V chunks (32 keys) are double-buffered in LDS via async DMA; softmax uses
// the S^T/O^T trick so all reductions are per-lane + one shfl_xor(16).
// ---------------------------------------------------------------------------
__global__ __launch_bounds__(128) void flash_attn_kernel(
    const _Float16* __restrict__ Qh, const _Float16* __restrict__ Kh,
    const _Float16* __restrict__ Vt, const int* __restrict__ mask,
    _Float16* __restrict__ att) {
  constexpr int KST = 72;  // K row stride in halfs (144B, conflict-free)
  constexpr int VST = 40;  // V^T row stride in halfs (80B, conflict-free)
  __shared__ __align__(16) _Float16 kbuf[2][32 * KST];
  __shared__ __align__(16) _Float16 vbuf[2][64 * VST];

  const int tid  = threadIdx.x;
  const int lane = tid & 31;
  const int wid  = tid >> 5;
  const int l15  = lane & 15;
  const int hi   = lane >> 4;
  const int bh   = blockIdx.y;
  const int b    = bh >> 4;
  const int h    = bh & (Hh - 1);
  const int q0   = (blockIdx.x * 4 + wid) * 16;

  // Q as B-fragments, with 1/sqrt(dh)=0.125 folded in (exact in f16)
  const _Float16* qrow = Qh + (size_t)(b * Ll + q0 + l15) * Dd + h * DH;
  v16h qb0 = *(const v16h*)(qrow + hi * 16);
  v16h qb1 = *(const v16h*)(qrow + 32 + hi * 16);
#pragma unroll
  for (int r = 0; r < 16; ++r) {
    qb0[r] = qb0[r] * (_Float16)0.125f;
    qb1[r] = qb1[r] * (_Float16)0.125f;
  }

  const _Float16* Kbase = Kh + (size_t)b * Ll * Dd + h * DH;
  const _Float16* Vbase = Vt + (size_t)bh * DH * Ll;
  const int* mrow = mask + b * Ll;

  // Cooperative async stage of one 32-key chunk (K: 32x64h, V^T: 64x32h)
  auto issue_chunk = [&](int kcn, int bufIdx) {
#pragma unroll
    for (int j = 0; j < 2; ++j) {  // K: 256 x 16B pieces over 128 threads
      const int p = tid + 128 * j;
      const int row = p >> 3, c8 = (p & 7) * 8;
      cp16_async(&kbuf[bufIdx][row * KST + c8],
                 Kbase + (size_t)(kcn + row) * Dd + c8);
    }
#pragma unroll
    for (int j = 0; j < 2; ++j) {  // V: 256 x 16B pieces
      const int p = tid + 128 * j;
      const int d = p >> 2, c8 = (p & 3) * 8;
      cp16_async(&vbuf[bufIdx][d * VST + c8],
                 Vbase + (size_t)d * Ll + kcn + c8);
    }
  };

  v8f o0 = {}, o1 = {}, o2 = {}, o3 = {};
  float mrun = -INFINITY, lrun = 0.f;

  issue_chunk(0, 0);
  int cur = 0;

  for (int kc = 0; kc < Ll; kc += 32) {
    const int kcn = (kc + 32 < Ll) ? kc + 32 : kc;  // last prefetch is benign
    issue_chunk(kcn, cur ^ 1);
    wait_async_le4();       // chunk for buf[cur] complete (in-order ASYNCcnt)
    __syncthreads();        // visible to all 4 waves

    // mask bits for keys kc..kc+31 (independent global load)
    const int mv = mrow[kc + lane];
    const unsigned mb32 = (unsigned)__ballot(mv != 0);
    // per-lane 8-bit windows: tile0 keys (hi*8)+0..7, tile1 keys 16+(hi*8)+0..7
    const unsigned mybits0 = (mb32 >> (hi * 8)) & 0xffu;
    const unsigned mybits1 = (mb32 >> (16 + hi * 8)) & 0xffu;

    // S^T = K * Q^T : two 16-key tiles from LDS
    v8f s0 = {}, s1 = {};
    {
      const _Float16* kr = &kbuf[cur][l15 * KST + hi * 16];
      V16U ka, kb2;
      ka.h[0]  = *(const v8h*)(kr);
      ka.h[1]  = *(const v8h*)(kr + 8);
      kb2.h[0] = *(const v8h*)(kr + 32);
      kb2.h[1] = *(const v8h*)(kr + 40);
      s0 = WMMA_F32_F16(ka.v, qb0, s0);
      s0 = WMMA_F32_F16(kb2.v, qb1, s0);
    }
    {
      const _Float16* kr = &kbuf[cur][(16 + l15) * KST + hi * 16];
      V16U ka, kb2;
      ka.h[0]  = *(const v8h*)(kr);
      ka.h[1]  = *(const v8h*)(kr + 8);
      kb2.h[0] = *(const v8h*)(kr + 32);
      kb2.h[1] = *(const v8h*)(kr + 40);
      s1 = WMMA_F32_F16(ka.v, qb0, s1);
      s1 = WMMA_F32_F16(kb2.v, qb1, s1);
    }

    // V^T fragments (issue ds loads early; consumed after softmax)
    V16U va0, va1, va2, va3;
    {
      const _Float16* vr = &vbuf[cur][l15 * VST + hi * 8];
      va0.h[0] = *(const v8h*)(vr);
      va0.h[1] = *(const v8h*)(vr + 16);
      vr += 16 * VST;
      va1.h[0] = *(const v8h*)(vr);
      va1.h[1] = *(const v8h*)(vr + 16);
      vr += 16 * VST;
      va2.h[0] = *(const v8h*)(vr);
      va2.h[1] = *(const v8h*)(vr + 16);
      vr += 16 * VST;
      va3.h[0] = *(const v8h*)(vr);
      va3.h[1] = *(const v8h*)(vr + 16);
    }

    // mask (constant-bit tests) + per-query running max (per lane-pair)
    float p0[8], p1[8];
    float mloc = -INFINITY;
#pragma unroll
    for (int r = 0; r < 8; ++r) {
      float x0 = s0[r];
      float x1 = s1[r];
      if (mybits0 & (1u << r)) x0 = -1e9f;
      if (mybits1 & (1u << r)) x1 = -1e9f;
      p0[r] = x0; p1[r] = x1;
      mloc = fmaxf(mloc, fmaxf(x0, x1));
    }
    mloc = fmaxf(mloc, __shfl_xor(mloc, 16, 32));
    const float mnew = fmaxf(mrun, mloc);
    const float c = __expf(mrun - mnew);
    float ssum = 0.f;
#pragma unroll
    for (int r = 0; r < 8; ++r) {
      p0[r] = __expf(p0[r] - mnew);
      p1[r] = __expf(p1[r] - mnew);
      ssum += p0[r] + p1[r];
    }
    ssum += __shfl_xor(ssum, 16, 32);
    lrun = fmaf(lrun, c, ssum);
    mrun = mnew;
#pragma unroll
    for (int r = 0; r < 8; ++r) {
      o0[r] *= c; o1[r] *= c; o2[r] *= c; o3[r] *= c;
    }

    // P (32 keys x 16 queries) as B-fragment: own-lane values, no lane swap
    V16U pb;
#pragma unroll
    for (int r = 0; r < 8; ++r) {
      pb.v[r]     = (_Float16)p0[r];
      pb.v[r + 8] = (_Float16)p1[r];
    }

    // O^T += V^T * P
    o0 = WMMA_F32_F16(va0.v, pb.v, o0);
    o1 = WMMA_F32_F16(va1.v, pb.v, o1);
    o2 = WMMA_F32_F16(va2.v, pb.v, o2);
    o3 = WMMA_F32_F16(va3.v, pb.v, o3);

    __syncthreads();  // all waves done reading buf[cur] before it is refilled
    cur ^= 1;
  }

  // epilogue: att[(b*L + q)*D + h*DH + d], normalized
  const float inv = 1.f / lrun;
  _Float16* arow = att + (size_t)(b * Ll + q0 + l15) * Dd + h * DH;
  v8h pk;
#pragma unroll
  for (int r = 0; r < 8; ++r) pk[r] = (_Float16)(o0[r] * inv);
  *(v8h*)(arow + 0 * 16 + hi * 8) = pk;
#pragma unroll
  for (int r = 0; r < 8; ++r) pk[r] = (_Float16)(o1[r] * inv);
  *(v8h*)(arow + 1 * 16 + hi * 8) = pk;
#pragma unroll
  for (int r = 0; r < 8; ++r) pk[r] = (_Float16)(o2[r] * inv);
  *(v8h*)(arow + 2 * 16 + hi * 8) = pk;
#pragma unroll
  for (int r = 0; r < 8; ++r) pk[r] = (_Float16)(o3[r] * inv);
  *(v8h*)(arow + 3 * 16 + hi * 8) = pk;
}

// ---------------------------------------------------------------------------
extern "C" void kernel_launch(void* const* d_in, const int* in_sizes, int n_in,
                              void* d_out, int out_size, void* d_ws,
                              size_t ws_size, hipStream_t stream) {
  const float* x    = (const float*)d_in[0];
  const int*   mask = (const int*)d_in[1];
  const float* wq = (const float*)d_in[2];
  const float* bq = (const float*)d_in[3];
  const float* wk = (const float*)d_in[4];
  const float* bk = (const float*)d_in[5];
  const float* wv = (const float*)d_in[6];
  const float* bv = (const float*)d_in[7];
  const float* wo = (const float*)d_in[8];
  const float* bo = (const float*)d_in[9];

  // Workspace carve (needs 40 MB). att aliases xh (x dead after QKV GEMMs).
  char* ws = (char*)d_ws;
  _Float16* xh  = (_Float16*)(ws + 0);                       // 16 MB
  _Float16* wqh = (_Float16*)(ws + (16u << 20));             // 2 MB each
  _Float16* wkh = (_Float16*)(ws + (18u << 20));
  _Float16* wvh = (_Float16*)(ws + (20u << 20));
  _Float16* woh = (_Float16*)(ws + (22u << 20));
  _Float16* Vt  = (_Float16*)(ws + (24u << 20));             // 16 MB
  _Float16* att = xh;                                        // alias
  // Q/K f16 live in d_out (exactly 32 MB), overwritten by final projection.
  _Float16* Qh = (_Float16*)d_out;
  _Float16* Kh = (_Float16*)d_out + (size_t)Mm * Dd;

  const int nX = Mm * Dd, nW = Dd * Dd;
  cvt_f32_to_f16_kernel<<<512, 256, 0, stream>>>(x, xh, nX);
  cvt_f32_to_f16_kernel<<<256, 256, 0, stream>>>(wq, wqh, nW);
  cvt_f32_to_f16_kernel<<<256, 256, 0, stream>>>(wk, wkh, nW);
  cvt_f32_to_f16_kernel<<<256, 256, 0, stream>>>(wv, wvh, nW);
  cvt_f32_to_f16_kernel<<<256, 256, 0, stream>>>(wo, woh, nW);

  dim3 ggrid(Dd / 128, Mm / 64);  // (8, 128): 64x128 block tile, 4 waves
  gemm_wmma_kernel<0><<<ggrid, 128, 0, stream>>>(xh, wqh, bq, Qh, Mm, Dd, Dd);
  gemm_wmma_kernel<0><<<ggrid, 128, 0, stream>>>(xh, wkh, bk, Kh, Mm, Dd, Dd);
  gemm_wmma_kernel<1><<<ggrid, 128, 0, stream>>>(xh, wvh, bv, Vt, Mm, Dd, Dd);

  dim3 agrid(Ll / 64, Bb * Hh);  // (32, 64): 4 waves/block, 16 queries/wave
  flash_attn_kernel<<<agrid, 128, 0, stream>>>(Qh, Kh, Vt, mask, att);

  gemm_wmma_kernel<2><<<ggrid, 128, 0, stream>>>(att, woh, bo, (float*)d_out,
                                                 Mm, Dd, Dd);
}
